// GenerativeROUND_38216619000175
// MI455X (gfx1250) — matrix operand order repaired and checked by reading, not compiled, not run on hardware
//
#include <hip/hip_runtime.h>

// Problem constants (match reference)
#define B_    64
#define S_    512
#define D_    17
#define H_    1024
#define C_    17
#define F_    (4 * H_)      // 4096 features
#define NPAD  32            // C padded to 32 for two 16-wide WMMA tiles

// LDS staging of B: pad each row by 8 halves (16B) to rotate bank groups
#define LDSROW (F_ + 8)     // 4104 halves = 8208 bytes per row
#define LDSBYTES (NPAD * LDSROW * 2)   // 262,656 bytes (fits 320KB WGP LDS)

#define PERIOD_F     12.566370614359172f   // 4*pi
#define TWO_PI_F     6.283185307179586f
#define INV_TWO_PI_F 0.15915494309189535f
#define PHASE_WELL_F 0.1f

typedef __attribute__((ext_vector_type(16))) _Float16 v16h;
typedef __attribute__((ext_vector_type(8)))  _Float16 v8h;
typedef __attribute__((ext_vector_type(8)))  float    v8f;

// ---------------------------------------------------------------------------
// Kernel 1: pack r_w [17,4096] f32 -> padded f16 [32,4096] (rows 17..31 = 0)
// ---------------------------------------------------------------------------
__global__ void pack_rw_kernel(const float* __restrict__ r_w,
                               _Float16* __restrict__ rwp) {
    int i = blockIdx.x * blockDim.x + threadIdx.x;   // 0 .. 32*4096-1
    int r = i / F_;
    int k = i - r * F_;
    float v = (r < C_) ? r_w[r * F_ + k] : 0.0f;
    rwp[i] = (_Float16)v;
}

// ---------------------------------------------------------------------------
// Kernel 2: fused embed + phase scan + feature expansion.
// One thread per (b,h) chain; loops s=0..511 sequentially (true dependency).
// Writes ph_seq (f32, output #2) and f16 features [M=B*S, F=4096] row-major.
// ---------------------------------------------------------------------------
__global__ void __launch_bounds__(256) scan_kernel(
    const float* __restrict__ x,     // [B,S,D]
    const float* __restrict__ e_w,   // [H,D]
    const float* __restrict__ e_b,   // [H]
    float* __restrict__ ph_out,      // [B,S,H]
    _Float16* __restrict__ feats)    // [B*S, 4H] f16
{
    int g = blockIdx.x * blockDim.x + threadIdx.x;   // b*H + h
    int b = g >> 10;            // / H_
    int h = g & (H_ - 1);

    // embed row for this h lives in registers (17 floats)
    float ew[D_];
#pragma unroll
    for (int d = 0; d < D_; ++d) ew[d] = e_w[h * D_ + d];
    float eb = e_b[h];

    const float* xrow = x + (size_t)b * S_ * D_;     // wave-uniform base
    float ph = 0.0f;

    for (int s = 0; s < S_; ++s) {
        // pt = <x[b,s,:], e_w[h,:]> + e_b[h]   (x loads are lane-uniform)
        float pt = eb;
#pragma unroll
        for (int d = 0; d < D_; ++d) pt = fmaf(xrow[d], ew[d], pt);
        xrow += D_;

        // delta = atan2(sin pt, cos pt) == wrap pt to (-pi, pi]
        float delta = fmaf(-TWO_PI_F, __builtin_rintf(pt * INV_TWO_PI_F), pt);

        // ph <- mod(ph + delta, 4*pi); range of (ph+delta) is (-P, 2P)
        float t = ph + delta;
        t = (t <  0.0f)     ? t + PERIOD_F : t;
        t = (t >= PERIOD_F) ? t - PERIOD_F : t;

        // phase-well snap
        ph = fmaf(-PHASE_WELL_F, __sinf(17.0f * t), t);

        int m = b * S_ + s;
        ph_out[(size_t)m * H_ + h] = ph;             // coalesced across h

        float s1, c1, s2, c2;
        __sincosf(ph, &s1, &c1);
        __sincosf(0.5f * ph, &s2, &c2);

        _Float16* fr = feats + (size_t)m * F_;       // coalesced 2B stores
        fr[h]          = (_Float16)c1;
        fr[H_ + h]     = (_Float16)s1;
        fr[2 * H_ + h] = (_Float16)c2;
        fr[3 * H_ + h] = (_Float16)s2;
    }
}

// ---------------------------------------------------------------------------
// Kernel 3: readout GEMM  logits[m,c] = feats[m,:] . r_w[c,:] + r_b[c]
// via V_WMMA_F32_16X16X32_F16.
//   - Whole padded B (32x4096 f16, 256KB) staged once per block into LDS
//     (CDNA5 WGP has 320KB LDS); K-loop reads B via ds_load, so the vector-
//     memory path carries only the compulsory A stream.
//   - One wave per 16-row M tile, both N tiles (cols 0..15, 16..31) sharing A.
// C/D layout per CDNA5 ISA 7.12.2; A and B share contiguous k order (k = 16h + s).
// ---------------------------------------------------------------------------
__global__ void __launch_bounds__(256) readout_wmma_kernel(
    const _Float16* __restrict__ feats,  // [32768, 4096]
    const _Float16* __restrict__ rwp,    // [32, 4096] padded
    const float* __restrict__ r_b,       // [17]
    float* __restrict__ logits)          // [B*S, 17]
{
    extern __shared__ _Float16 bsh[];    // [32][LDSROW] padded B in LDS

    // ---- cooperative B fill: 32 rows x 4096 halves, 16B chunks ----
    // 32*512 = 16384 chunks; 256 threads -> 64 chunks each
    for (int idx = threadIdx.x; idx < NPAD * (F_ / 8); idx += 256) {
        int r = idx >> 9;                // / (4096/8)
        int c = (idx & 511) << 3;        // half offset within row
        *(v8h*)(bsh + r * LDSROW + c) = *(const v8h*)(rwp + r * F_ + c);
    }
    __syncthreads();

    const int lane  = threadIdx.x & 31;
    const int wave  = (blockIdx.x * (blockDim.x >> 5)) + (threadIdx.x >> 5);
    const int mtile = wave;                       // 0..2047
    const int lm    = lane & 15;

    // A (16x32, MxK): lanes 0-15 hold K 0..15; lanes 16-31 hold K 16..31
    const int aoff = (lane < 16) ? 0 : 16;
    // B (32x16, KxN): lanes 0-15 hold K 0..15; lanes 16-31 hold K 16..31
    const int boff = (lane < 16) ? 0 : 16;

    const _Float16* arow = feats + (size_t)(mtile * 16 + lm) * F_;
    const _Float16* bl0  = bsh + lm * LDSROW + boff;          // N tile 0 rows
    const _Float16* bl1  = bsh + (lm + 16) * LDSROW + boff;   // N tile 1 rows

    v8f acc0 = {};
    v8f acc1 = {};

    for (int k0 = 0; k0 < F_; k0 += 32) {
        union { v16h v; v8h h[2]; } a;
        a.h[0] = *(const v8h*)(arow + k0 + aoff);        // VGPR0-3
        a.h[1] = *(const v8h*)(arow + k0 + aoff + 8);    // VGPR4-7
        v16h b0 = *(const v16h*)(bl0 + k0);              // ds_load from LDS
        v16h b1 = *(const v16h*)(bl1 + k0);

        acc0 = __builtin_amdgcn_wmma_f32_16x16x32_f16(
                   false, a.v, false, b0, (short)0, acc0, false, false);
        acc1 = __builtin_amdgcn_wmma_f32_16x16x32_f16(
                   false, a.v, false, b1, (short)0, acc1, false, false);
    }

    // C/D layout: VGPR v, lanes 0-15 -> M=v, N=lane; lanes 16-31 -> M=v+8
    const int mhi  = (lane >= 16) ? 8 : 0;
    const int col0 = lm;            // < 16 < 17: always valid
    const int col1 = 16 + lm;       // only col1 == 16 valid
    const float bias0 = r_b[col0];
#pragma unroll
    for (int v = 0; v < 8; ++v) {
        int m = mtile * 16 + v + mhi;
        logits[(size_t)m * C_ + col0] = acc0[v] + bias0;
        if (col1 < C_)
            logits[(size_t)m * C_ + col1] = acc1[v] + r_b[col1];
    }
}

// ---------------------------------------------------------------------------
extern "C" void kernel_launch(void* const* d_in, const int* in_sizes, int n_in,
                              void* d_out, int out_size, void* d_ws, size_t ws_size,
                              hipStream_t stream) {
    const float* x   = (const float*)d_in[0];   // [B,S,D]
    const float* e_w = (const float*)d_in[1];   // [H,D]
    const float* e_b = (const float*)d_in[2];   // [H]
    const float* r_w = (const float*)d_in[3];   // [C,4H]
    const float* r_b = (const float*)d_in[4];   // [C]

    // d_out = logits [B,S,C] then ph_seq [B,S,H], concatenated flat
    float* logits = (float*)d_out;
    float* ph_out = logits + (size_t)B_ * S_ * C_;

    // workspace layout: [0, 256KB) padded f16 r_w; then f16 features
    _Float16* rwp   = (_Float16*)d_ws;
    _Float16* feats = (_Float16*)((char*)d_ws + (size_t)NPAD * F_ * sizeof(_Float16));

    // 1) pack/pad r_w to f16 [32,4096]
    pack_rw_kernel<<<(NPAD * F_) / 256, 256, 0, stream>>>(r_w, rwp);

    // 2) fused embed + phase scan + feature expansion: B*H = 65536 threads
    scan_kernel<<<(B_ * H_) / 256, 256, 0, stream>>>(x, e_w, e_b, ph_out, feats);

    // 3) WMMA readout: 2048 M-tiles, 1 wave each, 8 waves per 256-thread block,
    //    257KB dynamic LDS for the staged B matrix
    readout_wmma_kernel<<<(B_ * S_ / 16) / 8, 256, LDSBYTES, stream>>>(
        feats, rwp, r_b, logits);
}
